// Head_55482387530086
// MI455X (gfx1250) — matrix operand order repaired and checked by reading, not compiled, not run on hardware
//
#include <hip/hip_runtime.h>

#define BATCH 8
#define TSEQ  2048
#define NE    2048
#define HD    128
#define BT    (BATCH * TSEQ)

typedef __attribute__((ext_vector_type(16))) __bf16 v16bf;
typedef __attribute__((ext_vector_type(8)))  float  v8f;
typedef __attribute__((ext_vector_type(4)))  unsigned int v4u;
typedef __attribute__((ext_vector_type(8)))  int    v8i;
typedef __attribute__((ext_vector_type(4)))  int    v4i;

union B16Frag { unsigned u[8]; v16bf v; };

#if defined(__gfx1250__) && __has_builtin(__builtin_amdgcn_tensor_load_to_lds)
#define USE_TDM 1
#else
#define USE_TDM 0
#endif

__device__ __forceinline__ unsigned short bf16bits(float f) {
    union { __bf16 h; unsigned short s; } t;
    t.h = (__bf16)f;
    return t.s;
}
__device__ __forceinline__ unsigned pack2(float a, float b) {
    return (unsigned)bf16bits(a) | ((unsigned)bf16bits(b) << 16);
}
// A-fragment K index for 16-bit 16x32 A matrix (ISA 7.12.2)
__device__ __forceinline__ int a_kpat(int v, int half) {
    return (v < 4) ? (2 * v + 8 * half) : (16 + 2 * (v - 4) + 8 * half);
}

#if USE_TDM
// Issue a 2D TDM tile load: tile [tile1 rows x tile0 elems] of 2-byte elements,
// row stride = stride0 elems, into contiguous LDS at lds_off.
// D# layout per CDNA5 ISA 8.3 (group0) / 8.4 (group1).
__device__ __forceinline__ void tdm_load_2d(unsigned lds_off, unsigned long long gaddr,
                                            unsigned dim0, unsigned dim1,
                                            unsigned tile0, unsigned tile1,
                                            unsigned stride0)
{
    v4u g0;
    g0[0] = 1u;                                            // count=1, load, no gather
    g0[1] = lds_off;                                       // lds_addr
    g0[2] = (unsigned)gaddr;                               // global_addr[31:0]
    g0[3] = ((unsigned)(gaddr >> 32) & 0x01FFFFFFu) | (2u << 30); // addr[56:32] | type=2
    v8i g1;
    g1[0] = (int)(1u << 16);                               // data_size=1 (2 bytes)
    g1[1] = (int)(dim0 << 16);                             // tensor_dim0[15:0]
    g1[2] = (int)((dim0 >> 16) | (dim1 << 16));            // dim0[31:16] | dim1[15:0]
    g1[3] = (int)((dim1 >> 16) | (tile0 << 16));           // dim1[31:16] | tile_dim0
    g1[4] = (int)(tile1 & 0xffffu);                        // tile_dim1 (tile_dim2=0)
    g1[5] = (int)stride0;                                  // tensor_dim0_stride[31:0]
    g1[6] = 0;
    g1[7] = 0;
    v4i z4 = {};
#if __clang_major__ >= 23
    v8i z8 = {};
    __builtin_amdgcn_tensor_load_to_lds(g0, g1, z4, z4, z8, 0);
#else
    __builtin_amdgcn_tensor_load_to_lds(g0, g1, z4, z4, 0);
#endif
}
#endif

// --------------------------------------------------------------------------
// Kernel 1: QKV projection.  x[BT,NE] (f32) x W[NE,HD] (f32) -> bf16
// Q,K stored [BT][HD]; V stored transposed vT[b][d][TSEQ] (free, enables TDM
// tile loads of V in kernel 2 without an LDS transpose).
// grid = (BT/16, 3), block = 256 (8 waves; wave w owns cols [16w,16w+16))
// --------------------------------------------------------------------------
__global__ __launch_bounds__(256) void qkv_proj_kernel(
    const float* __restrict__ x,
    const float* __restrict__ Wq,
    const float* __restrict__ Wk,
    const float* __restrict__ Wv,
    unsigned short* __restrict__ qkv)
{
    const int mt    = blockIdx.x;
    const int which = blockIdx.y;         // 0=Q 1=K 2=V
    const float* W  = (which == 0) ? Wq : ((which == 1) ? Wk : Wv);
    unsigned short* dst = qkv + (size_t)which * (size_t)BT * HD;

    __shared__ __align__(16) unsigned xs[16 * 16];        // 16x32 bf16 as u32 pairs
    __shared__ __align__(16) unsigned short wt[HD * 32];  // W chunk transposed [col][k]

    const int tid  = threadIdx.x;
    const int lane = tid & 31;
    const int w    = tid >> 5;
    const int mrow = lane & 15;
    const int half = lane >> 4;

    v8f acc = {};

    for (int kc = 0; kc < NE / 32; ++kc) {
        __syncthreads();
        // stage x chunk 16x32 f32 -> bf16 packed (1 u32 per thread)
        {
            int r  = tid >> 4;
            int c2 = tid & 15;
            const float* xp = x + ((size_t)(mt * 16 + r)) * NE + kc * 32 + c2 * 2;
            xs[r * 16 + c2] = pack2(xp[0], xp[1]);
            if (kc + 1 < NE / 32) __builtin_prefetch(xp + 32, 0, 0);
        }
        // stage W chunk 32x128 f32 -> bf16 transposed [col][k]
        #pragma unroll
        for (int it = 0; it < 8; ++it) {
            int idx = it * 256 + tid;
            int k   = idx >> 6;
            int c2  = idx & 63;
            const float* wp = W + ((size_t)(kc * 32 + k)) * HD + c2 * 2;
            wt[(c2 * 2 + 0) * 32 + k] = bf16bits(wp[0]);
            wt[(c2 * 2 + 1) * 32 + k] = bf16bits(wp[1]);
            if (it == 0 && kc + 1 < NE / 32) __builtin_prefetch(wp + 32 * HD, 0, 0);
        }
        __syncthreads();

        B16Frag A, Bm;
        #pragma unroll
        for (int v = 0; v < 8; ++v)
            A.u[v] = xs[mrow * 16 + (a_kpat(v, half) >> 1)];
        const unsigned* wt32 = (const unsigned*)wt;
        const int col = w * 16 + mrow;
        #pragma unroll
        for (int v = 0; v < 8; ++v) {
            int k = a_kpat(v, half);
            Bm.u[v] = wt32[col * 16 + (k >> 1)];
        }
        acc = __builtin_amdgcn_wmma_f32_16x16x32_bf16(
                  false, A.v, false, Bm.v, (short)0, acc, false, false);
    }

    #pragma unroll
    for (int r = 0; r < 8; ++r) {
        int row = mt * 16 + r + 8 * half;        // global token index bt
        int col = w * 16 + mrow;
        unsigned short bits = bf16bits(acc[r]);
        if (which == 2) {
            int bb = row >> 11;                  // row / TSEQ
            int tt = row & (TSEQ - 1);
            dst[((size_t)bb * HD + col) * TSEQ + tt] = bits;   // vT[b][d][t]
        } else {
            dst[(size_t)row * HD + col] = bits;
        }
    }
}

// --------------------------------------------------------------------------
// Kernel 2: causal flash attention over bf16 Q,K,vT -> f32 out
// grid = (TSEQ/64, BATCH), block = 128 (4 waves; wave w owns 16 query rows)
// K/V tiles double-buffered in LDS; Tensor Data Mover prefetches tile jb+1
// while all waves compute on tile jb (deferred s_wait_tensorcnt).
// --------------------------------------------------------------------------
__global__ __launch_bounds__(128) void flash_attn_kernel(
    const unsigned short* __restrict__ qkv,
    float* __restrict__ out)
{
    const int qb   = blockIdx.x;
    const int b    = blockIdx.y;
    const int tid  = threadIdx.x;
    const int lane = tid & 31;
    const int w    = tid >> 5;
    const int mrow = lane & 15;
    const int half = lane >> 4;
    const int qbase = qb * 64;
    const int qrow0 = qbase + w * 16;

    const size_t plane = (size_t)BT * HD;
    const unsigned short* Qp  = qkv;
    const unsigned short* Kp  = qkv + plane;
    const unsigned short* Vtp = qkv + 2 * plane;   // vT[b][d][T]
    const unsigned* Q32  = (const unsigned*)Qp  + ((size_t)b * TSEQ) * (HD / 2);
    const unsigned* K32  = (const unsigned*)Kp  + ((size_t)b * TSEQ) * (HD / 2);
    const unsigned* Vt32 = (const unsigned*)Vtp + ((size_t)b * HD) * (TSEQ / 2);
    (void)K32; (void)Vt32;

    __shared__ __align__(16) unsigned       ks4[2][32 * (HD / 2)];  // K tiles [key][dimpair]
    __shared__ __align__(16) unsigned short vt[2][HD * 32];         // V tiles [dim][key]
    __shared__ __align__(16) unsigned short pw[4][16 * 32];         // per-wave P tile

    // Q fragments for this wave's 16 rows (A layout, K = 4 chunks of 32)
    B16Frag qf[4];
    #pragma unroll
    for (int c = 0; c < 4; ++c)
        #pragma unroll
        for (int v = 0; v < 8; ++v) {
            int k = 32 * c + a_kpat(v, half);
            qf[c].u[v] = Q32[(size_t)(qrow0 + mrow) * (HD / 2) + (k >> 1)];
        }

    float mrun[8], lrun[8];
    v8f acc[8];
    #pragma unroll
    for (int r = 0; r < 8; ++r) { mrun[r] = -3.0e38f; lrun[r] = 0.0f; }
    #pragma unroll
    for (int nt = 0; nt < 8; ++nt) acc[nt] = (v8f){};

    const int nkv = qbase / 32 + 2;
    const float sc = 0.08838834764831845f;   // 1/sqrt(128)

#if USE_TDM
    // prologue: DMA tile 0 into buffer 0
    if (w == 0) {
        tdm_load_2d((unsigned)(unsigned long long)(ks4[0]),
                    (unsigned long long)(const void*)(Kp + ((size_t)b * TSEQ) * HD),
                    HD, 32, HD, 32, HD);
        tdm_load_2d((unsigned)(unsigned long long)(vt[0]),
                    (unsigned long long)(const void*)(Vtp + ((size_t)b * HD) * TSEQ),
                    32, HD, 32, HD, TSEQ);
    }
#else
    {
        const unsigned* kp = K32;
        unsigned* vt32w = (unsigned*)vt[0];
        #pragma unroll
        for (int i = 0; i < 16; ++i) {
            int idx = i * 128 + tid;
            ks4[0][idx] = kp[idx];
            int d = idx >> 4, kk = idx & 15;
            vt32w[idx] = Vt32[(size_t)d * (TSEQ / 2) + kk];
        }
    }
#endif

    for (int jb = 0; jb < nkv; ++jb) {
        const int cur = jb & 1;
        __syncthreads();                 // all waves done reading buffer !cur
#if USE_TDM
        if (w == 0) {
            __builtin_amdgcn_s_wait_tensorcnt(0);   // tiles for jb have landed
            if (jb + 1 < nkv) {
                int jn = jb + 1;
                tdm_load_2d((unsigned)(unsigned long long)(ks4[cur ^ 1]),
                            (unsigned long long)(const void*)(Kp + ((size_t)b * TSEQ + jn * 32) * HD),
                            HD, 32, HD, 32, HD);
                tdm_load_2d((unsigned)(unsigned long long)(vt[cur ^ 1]),
                            (unsigned long long)(const void*)(Vtp + ((size_t)b * HD) * TSEQ + jn * 32),
                            32, HD, 32, HD, TSEQ);
            }
        }
#else
        if (jb + 1 < nkv) {
            int jn = jb + 1;
            const unsigned* kp = K32 + (size_t)(jn * 32) * (HD / 2);
            unsigned* vt32w = (unsigned*)vt[cur ^ 1];
            #pragma unroll
            for (int i = 0; i < 16; ++i) {
                int idx = i * 128 + tid;
                ks4[cur ^ 1][idx] = kp[idx];
                int d = idx >> 4, kk = idx & 15;
                vt32w[idx] = Vt32[(size_t)d * (TSEQ / 2) + jn * 16 + kk];
            }
        }
#endif
        __syncthreads();                 // buffer cur now readable by all waves

        if (jb * 32 > qrow0 + 15) continue;   // wave-uniform causal skip

        // S = Q K^T for 16 rows x 32 keys (two 16x16 tiles)
        v8f s0 = (v8f){}, s1 = (v8f){};
        #pragma unroll
        for (int c = 0; c < 4; ++c) {
            B16Frag kb0, kb1;
            #pragma unroll
            for (int v = 0; v < 8; ++v) {
                int k = 32 * c + a_kpat(v, half);
                kb0.u[v] = ks4[cur][(0  + mrow) * (HD / 2) + (k >> 1)];
                kb1.u[v] = ks4[cur][(16 + mrow) * (HD / 2) + (k >> 1)];
            }
            s0 = __builtin_amdgcn_wmma_f32_16x16x32_bf16(
                     false, qf[c].v, false, kb0.v, (short)0, s0, false, false);
            s1 = __builtin_amdgcn_wmma_f32_16x16x32_bf16(
                     false, qf[c].v, false, kb1.v, (short)0, s1, false, false);
        }

        // scale + causal mask + online softmax (row stats in 16-lane halves)
        const int key0 = jb * 32 + mrow;
        const int key1 = key0 + 16;
        float corrs[8];
        #pragma unroll
        for (int r = 0; r < 8; ++r) {
            int qrow = qrow0 + r + 8 * half;
            float v0 = s0[r] * sc, v1 = s1[r] * sc;
            if (key0 > qrow) v0 = -3.0e38f;
            if (key1 > qrow) v1 = -3.0e38f;
            float mx = fmaxf(v0, v1);
            #pragma unroll
            for (int o = 8; o >= 1; o >>= 1) mx = fmaxf(mx, __shfl_xor(mx, o, 32));
            float mnew = fmaxf(mrun[r], mx);
            float corr = __expf(mrun[r] - mnew);
            mrun[r] = mnew;
            float p0 = __expf(v0 - mnew);
            float p1 = __expf(v1 - mnew);
            float rs = p0 + p1;
            #pragma unroll
            for (int o = 8; o >= 1; o >>= 1) rs += __shfl_xor(rs, o, 32);
            lrun[r] = lrun[r] * corr + rs;
            corrs[r] = corr;
            s0[r] = p0; s1[r] = p1;
        }
        #pragma unroll
        for (int nt = 0; nt < 8; ++nt)
            #pragma unroll
            for (int r = 0; r < 8; ++r) acc[nt][r] *= corrs[r];

        // P: C layout -> A layout via wave-private LDS transpose
        #pragma unroll
        for (int r = 0; r < 8; ++r) {
            int m = r + 8 * half;
            pw[w][m * 32 + mrow]      = bf16bits(s0[r]);
            pw[w][m * 32 + 16 + mrow] = bf16bits(s1[r]);
        }
        B16Frag pa;
        const unsigned* pw32 = (const unsigned*)pw[w];
        #pragma unroll
        for (int v = 0; v < 8; ++v)
            pa.u[v] = pw32[mrow * 16 + (a_kpat(v, half) >> 1)];

        // O += P * V
        const unsigned* vt32 = (const unsigned*)vt[cur];
        #pragma unroll
        for (int nt = 0; nt < 8; ++nt) {
            B16Frag vb;
            int dcol = nt * 16 + mrow;
            #pragma unroll
            for (int v = 0; v < 8; ++v) {
                int kv = a_kpat(v, half);
                vb.u[v] = vt32[dcol * 16 + (kv >> 1)];
            }
            acc[nt] = __builtin_amdgcn_wmma_f32_16x16x32_bf16(
                          false, pa.v, false, vb.v, (short)0, acc[nt], false, false);
        }
    }

    // epilogue: O = acc / l
    float* Ob = out + ((size_t)b * TSEQ) * HD;
    #pragma unroll
    for (int r = 0; r < 8; ++r) {
        float inv = 1.0f / lrun[r];
        int row = qrow0 + r + 8 * half;
        #pragma unroll
        for (int nt = 0; nt < 8; ++nt)
            Ob[(size_t)row * HD + nt * 16 + mrow] = acc[nt][r] * inv;
    }
}

// --------------------------------------------------------------------------
extern "C" void kernel_launch(void* const* d_in, const int* in_sizes, int n_in,
                              void* d_out, int out_size, void* d_ws, size_t ws_size,
                              hipStream_t stream) {
    const float* x  = (const float*)d_in[0];
    const float* Wq = (const float*)d_in[1];
    const float* Wk = (const float*)d_in[2];
    const float* Wv = (const float*)d_in[3];
    float* out = (float*)d_out;
    unsigned short* qkv = (unsigned short*)d_ws;   // 3 * BT * HD bf16 = 12.6 MB

    dim3 g1(BT / 16, 3);
    qkv_proj_kernel<<<g1, 256, 0, stream>>>(x, Wq, Wk, Wv, qkv);

    dim3 g2(TSEQ / 64, BATCH);
    flash_attn_kernel<<<g2, 128, 0, stream>>>(qkv, out);
}
